// MultiHeadAttention_5823975654052
// MI455X (gfx1250) — matrix operand order repaired and checked by reading, not compiled, not run on hardware
//
#include <hip/hip_runtime.h>
#include <hip/hip_bf16.h>

// MI455X / gfx1250 wave32 WMMA implementation of multi-head attention.
// All matmuls (QKV proj, Q·K^T, P·V, out proj) run on v_wmma_f32_16x16x32_bf16.

typedef __bf16 bf16_t;
typedef __attribute__((ext_vector_type(16))) __bf16 v16bf;  // 32B: A/B operand
typedef __attribute__((ext_vector_type(8)))  __bf16 v8bf;   // 16B chunk
typedef __attribute__((ext_vector_type(8)))  float  v8f;    // C/D accumulator

#define D_MODEL 512
#define SEQ     4096
#define NHEAD   8
#define DK      64
#define BATCH   2

// ---- helpers -------------------------------------------------------------

// Build a 16x32 bf16 A-operand chunk for one lane from an LDS row pointer.
// Contiguous k order used here (B reads match it): lane L (m = L&15, hf = L>>4):
//   VGPR j=0..3 -> K = hf*16 .. hf*16+7
//   VGPR j=4..7 -> K = hf*16+8 .. hf*16+15
// => two contiguous 8-element (16B) reads, matching contiguous B reads.
__device__ __forceinline__ v16bf load_a_chunk(const bf16_t* rowp, int hf) {
  union { v16bf v; v8bf h[2]; } u;
  u.h[0] = *(const v8bf*)(rowp + hf * 16);
  u.h[1] = *(const v8bf*)(rowp + hf * 16 + 8);
  return u.v;
}

__device__ __forceinline__ v8f zero8() {
  v8f z = {0.f, 0.f, 0.f, 0.f, 0.f, 0.f, 0.f, 0.f};
  return z;
}

// ---- kernel 1: fused Q/K/V projection  y = x @ W^T + b  -> bf16 heads ----
// C[m,n] = sum_k X[m,k] * W[n,k]; both contiguous in k => no transposes.
// WG tile: 128 (M) x 64 (N); 4 waves, each owns 32 rows (2 m-tiles) to reuse
// every B-operand across two WMMAs.
// Output layout: dst[((b*8+h)*4096 + s)*64 + d], bf16.
__global__ __launch_bounds__(128) void proj_qkv_kernel(
    const float* __restrict__ qi, const float* __restrict__ ki, const float* __restrict__ vi,
    const float* __restrict__ wq, const float* __restrict__ bq,
    const float* __restrict__ wk, const float* __restrict__ bk,
    const float* __restrict__ wv, const float* __restrict__ bv,
    bf16_t* __restrict__ qh, bf16_t* __restrict__ kh, bf16_t* __restrict__ vh)
{
  const float* X; const float* W; const float* bias; bf16_t* dst;
  if (blockIdx.z == 0)      { X = qi; W = wq; bias = bq; dst = qh; }
  else if (blockIdx.z == 1) { X = ki; W = wk; bias = bk; dst = kh; }
  else                      { X = vi; W = wv; bias = bv; dst = vh; }

  const int m0 = blockIdx.x * 128;
  const int n0 = blockIdx.y * 64;

  __shared__ __align__(32) bf16_t sX[128 * 32];
  __shared__ __align__(32) bf16_t sW[64 * 32];

  const int tid  = threadIdx.x;
  const int lane = tid & 31;
  const int wv32 = tid >> 5;       // wave 0..3: rows 32*wv32 .. +31
  const int hf   = lane >> 4;
  const int l16  = lane & 15;

  v8f acc[2][4];
#pragma unroll
  for (int mt = 0; mt < 2; ++mt)
#pragma unroll
    for (int nt = 0; nt < 4; ++nt) acc[mt][nt] = zero8();

  for (int k0 = 0; k0 < D_MODEL; k0 += 32) {
    // stage X rows (1 row / thread) and W rows (2 threads / row), f32 -> bf16
    {
      const float4* gx = (const float4*)(X + (size_t)(m0 + tid) * D_MODEL + k0);
      bf16_t* dx = sX + tid * 32;
#pragma unroll
      for (int j = 0; j < 8; ++j) {
        float4 f = gx[j];
        dx[j*4+0] = (bf16_t)f.x; dx[j*4+1] = (bf16_t)f.y;
        dx[j*4+2] = (bf16_t)f.z; dx[j*4+3] = (bf16_t)f.w;
      }
      const int wrow = tid >> 1, wseg = (tid & 1) * 16;
      const float4* gw = (const float4*)(W + (size_t)(n0 + wrow) * D_MODEL + k0 + wseg);
      bf16_t* dw = sW + wrow * 32 + wseg;
#pragma unroll
      for (int j = 0; j < 4; ++j) {
        float4 g = gw[j];
        dw[j*4+0] = (bf16_t)g.x; dw[j*4+1] = (bf16_t)g.y;
        dw[j*4+2] = (bf16_t)g.z; dw[j*4+3] = (bf16_t)g.w;
      }
    }
    __syncthreads();

    v16bf a[2];
    a[0] = load_a_chunk(sX + (wv32 * 32 + l16) * 32, hf);
    a[1] = load_a_chunk(sX + (wv32 * 32 + 16 + l16) * 32, hf);
#pragma unroll
    for (int nt = 0; nt < 4; ++nt) {
      // B 32x16: lane n = nt*16+l16, K = hf*16 + e  (contiguous 32B read)
      v16bf b = *(const v16bf*)(sW + (nt * 16 + l16) * 32 + hf * 16);
#pragma unroll
      for (int mt = 0; mt < 2; ++mt)
        acc[mt][nt] = __builtin_amdgcn_wmma_f32_16x16x32_bf16(
            false, a[mt], false, b, (short)0, acc[mt][nt], false, false);
    }
    __syncthreads();
  }

  // epilogue: + bias, write bf16 into head-major layout
#pragma unroll
  for (int nt = 0; nt < 4; ++nt) {
    int n = n0 + nt * 16 + l16;
    float bb = bias[n];
    int h = n >> 6, d = n & 63;
#pragma unroll
    for (int mt = 0; mt < 2; ++mt) {
#pragma unroll
      for (int i = 0; i < 8; ++i) {
        int m = m0 + wv32 * 32 + mt * 16 + hf * 8 + i;   // C row = i + hf*8
        int bI = m >> 12, s = m & 4095;
        dst[(((size_t)(bI * NHEAD + h)) * SEQ + s) * DK + d] =
            (bf16_t)(acc[mt][nt][i] + bb);
      }
    }
  }
}

// ---- kernel 2: flash attention -------------------------------------------
// One workgroup = one (b,h) x 64 query rows; 4 waves x 16 rows.
__global__ __launch_bounds__(128) void attn_kernel(
    const bf16_t* __restrict__ qh, const bf16_t* __restrict__ kh,
    const bf16_t* __restrict__ vh, bf16_t* __restrict__ attn)
{
  const int bh = blockIdx.y;            // b*8 + h
  const int q0 = blockIdx.x * 64;

  const int tid  = threadIdx.x;
  const int lane = tid & 31;
  const int wv32 = tid >> 5;
  const int hf   = lane >> 4;
  const int l16  = lane & 15;

  __shared__ __align__(32) bf16_t sQ [64 * 64];
  __shared__ __align__(32) bf16_t sK [64 * 64];
  __shared__ __align__(32) bf16_t sVt[64 * 64];   // V transposed: [d][t]
  __shared__ __align__(32) bf16_t sP [64 * 64];
  __shared__ float sS[64 * 64];
  __shared__ float row_m[64], row_l[64], row_scale[64], sRed[128];

  if (tid < 64) { row_m[tid] = -1e30f; row_l[tid] = 0.f; }

  // stage Q block (contiguous 8KB) and hoist A-operands into registers
  {
    const v8bf* g = (const v8bf*)(qh + ((size_t)bh * SEQ + q0) * DK);
    v8bf* sh = (v8bf*)sQ;
    for (int j = tid; j < 512; j += 128) sh[j] = g[j];
  }
  __syncthreads();
  v16bf aQ[2];
  {
    const bf16_t* rowp = sQ + (wv32 * 16 + l16) * DK;
    aQ[0] = load_a_chunk(rowp, hf);
    aQ[1] = load_a_chunk(rowp + 32, hf);
  }

  v8f o[4] = { zero8(), zero8(), zero8(), zero8() };

  for (int t0 = 0; t0 < SEQ; t0 += 64) {
    // gfx1250 prefetch of next K/V tile
    if (t0 + 64 < SEQ) {
      const bf16_t* nk = kh + ((size_t)bh * SEQ + t0 + 64) * DK;
      const bf16_t* nv = vh + ((size_t)bh * SEQ + t0 + 64) * DK;
      __builtin_prefetch(tid < 64 ? (const void*)(nk + tid * 64)
                                  : (const void*)(nv + (tid - 64) * 64), 0, 1);
    }
    // stage K row-major, V transposed
    {
      const v8bf* gK = (const v8bf*)(kh + ((size_t)bh * SEQ + t0) * DK);
      v8bf* shK = (v8bf*)sK;
      for (int j = tid; j < 512; j += 128) shK[j] = gK[j];

      int t = tid >> 1, dseg = (tid & 1) * 32;
      const v8bf* gV = (const v8bf*)(vh + ((size_t)bh * SEQ + t0 + t) * DK + dseg);
#pragma unroll
      for (int j = 0; j < 4; ++j) {
        v8bf val = gV[j];
#pragma unroll
        for (int e = 0; e < 8; ++e) sVt[(dseg + j * 8 + e) * 64 + t] = val[e];
      }
    }
    __syncthreads();

    // scores: S = (Q @ K^T) * 1/sqrt(64); write scaled f32 to sS
#pragma unroll
    for (int nt = 0; nt < 4; ++nt) {
      v8f c = zero8();
#pragma unroll
      for (int kc = 0; kc < 2; ++kc) {
        // B[k=d, n=t] = K[t,d] : contiguous d read from sK row t
        v16bf bK = *(const v16bf*)(sK + (nt * 16 + l16) * 64 + kc * 32 + hf * 16);
        c = __builtin_amdgcn_wmma_f32_16x16x32_bf16(
            false, aQ[kc], false, bK, (short)0, c, false, false);
      }
      const int col = nt * 16 + l16;
#pragma unroll
      for (int i = 0; i < 8; ++i)
        sS[(wv32 * 16 + hf * 8 + i) * 64 + col] = c[i] * 0.125f;
    }
    __syncthreads();

    // online softmax: 2 threads per row
    {
      int r = tid >> 1, hh = tid & 1;
      const float* srow = sS + r * 64 + hh * 32;
      float tmax = -1e30f;
#pragma unroll 8
      for (int t = 0; t < 32; ++t) tmax = fmaxf(tmax, srow[t]);
      sRed[tid] = tmax;
      __syncthreads();
      float tile_max = fmaxf(sRed[r * 2], sRed[r * 2 + 1]);
      float m_old = row_m[r];
      float m_new = fmaxf(m_old, tile_max);
      float psum = 0.f;
      bf16_t* prow = sP + r * 64 + hh * 32;
#pragma unroll 8
      for (int t = 0; t < 32; ++t) {
        float p = __expf(srow[t] - m_new);
        prow[t] = (bf16_t)p;
        psum += p;
      }
      __syncthreads();
      sRed[tid] = psum;
      __syncthreads();
      if (hh == 0) {
        float scale = __expf(m_old - m_new);
        row_scale[r] = scale;
        row_l[r] = row_l[r] * scale + sRed[r * 2] + sRed[r * 2 + 1];
        row_m[r] = m_new;
      }
    }
    __syncthreads();

    // rescale running O, then O += P @ V
    float fs[8];
#pragma unroll
    for (int i = 0; i < 8; ++i) fs[i] = row_scale[wv32 * 16 + hf * 8 + i];
#pragma unroll
    for (int nt = 0; nt < 4; ++nt)
#pragma unroll
      for (int i = 0; i < 8; ++i) o[nt][i] *= fs[i];

    const bf16_t* prowp = sP + (wv32 * 16 + l16) * 64;
    v16bf aP0 = load_a_chunk(prowp, hf);
    v16bf aP1 = load_a_chunk(prowp + 32, hf);
#pragma unroll
    for (int nt = 0; nt < 4; ++nt) {
      // B[k=t, n=d] = V[t,d] = sVt[d][t] : contiguous t read
      v16bf bV0 = *(const v16bf*)(sVt + (nt * 16 + l16) * 64 + hf * 16);
      o[nt] = __builtin_amdgcn_wmma_f32_16x16x32_bf16(
          false, aP0, false, bV0, (short)0, o[nt], false, false);
      v16bf bV1 = *(const v16bf*)(sVt + (nt * 16 + l16) * 64 + 32 + hf * 16);
      o[nt] = __builtin_amdgcn_wmma_f32_16x16x32_bf16(
          false, aP1, false, bV1, (short)0, o[nt], false, false);
    }
    __syncthreads();
  }

  // epilogue: O /= l, write bf16 into [B*S, 512] (head h at cols h*64..)
  float invl[8];
#pragma unroll
  for (int i = 0; i < 8; ++i) invl[i] = 1.0f / row_l[wv32 * 16 + hf * 8 + i];
  const int h = bh & 7, bI = bh >> 3;
#pragma unroll
  for (int nt = 0; nt < 4; ++nt) {
    int d = nt * 16 + l16;
#pragma unroll
    for (int i = 0; i < 8; ++i) {
      int s = q0 + wv32 * 16 + hf * 8 + i;
      attn[((size_t)(bI * SEQ + s)) * D_MODEL + h * DK + d] =
          (bf16_t)(o[nt][i] * invl[i]);
    }
  }
}

// ---- kernel 3: output projection  out = attn @ Wo^T + bo  (f32 out) ------
// Same 128x64 tiling as proj_qkv; X is already bf16.
__global__ __launch_bounds__(128) void proj_out_kernel(
    const bf16_t* __restrict__ X, const float* __restrict__ W,
    const float* __restrict__ bias, float* __restrict__ out)
{
  const int m0 = blockIdx.x * 128;
  const int n0 = blockIdx.y * 64;

  __shared__ __align__(32) bf16_t sX[128 * 32];
  __shared__ __align__(32) bf16_t sW[64 * 32];

  const int tid  = threadIdx.x;
  const int lane = tid & 31;
  const int wv32 = tid >> 5;
  const int hf   = lane >> 4;
  const int l16  = lane & 15;

  v8f acc[2][4];
#pragma unroll
  for (int mt = 0; mt < 2; ++mt)
#pragma unroll
    for (int nt = 0; nt < 4; ++nt) acc[mt][nt] = zero8();

  for (int k0 = 0; k0 < D_MODEL; k0 += 32) {
    // stage X (bf16 copy, 1 row / thread) and W (f32 -> bf16)
    {
      const v16bf* gx = (const v16bf*)(X + (size_t)(m0 + tid) * D_MODEL + k0);
      v16bf* dx = (v16bf*)(sX + tid * 32);
      dx[0] = gx[0];
      dx[1] = gx[1];

      const int wrow = tid >> 1, wseg = (tid & 1) * 16;
      const float4* gw = (const float4*)(W + (size_t)(n0 + wrow) * D_MODEL + k0 + wseg);
      bf16_t* dw = sW + wrow * 32 + wseg;
#pragma unroll
      for (int j = 0; j < 4; ++j) {
        float4 g = gw[j];
        dw[j*4+0] = (bf16_t)g.x; dw[j*4+1] = (bf16_t)g.y;
        dw[j*4+2] = (bf16_t)g.z; dw[j*4+3] = (bf16_t)g.w;
      }
    }
    __syncthreads();

    v16bf a[2];
    a[0] = load_a_chunk(sX + (wv32 * 32 + l16) * 32, hf);
    a[1] = load_a_chunk(sX + (wv32 * 32 + 16 + l16) * 32, hf);
#pragma unroll
    for (int nt = 0; nt < 4; ++nt) {
      v16bf b = *(const v16bf*)(sW + (nt * 16 + l16) * 32 + hf * 16);
#pragma unroll
      for (int mt = 0; mt < 2; ++mt)
        acc[mt][nt] = __builtin_amdgcn_wmma_f32_16x16x32_bf16(
            false, a[mt], false, b, (short)0, acc[mt][nt], false, false);
    }
    __syncthreads();
  }

#pragma unroll
  for (int nt = 0; nt < 4; ++nt) {
    int n = n0 + nt * 16 + l16;
    float bb = bias[n];
#pragma unroll
    for (int mt = 0; mt < 2; ++mt) {
#pragma unroll
      for (int i = 0; i < 8; ++i) {
        int m = m0 + wv32 * 32 + mt * 16 + hf * 8 + i;
        out[(size_t)m * D_MODEL + n] = acc[mt][nt][i] + bb;
      }
    }
  }
}

// ---- launcher ------------------------------------------------------------
extern "C" void kernel_launch(void* const* d_in, const int* in_sizes, int n_in,
                              void* d_out, int out_size, void* d_ws, size_t ws_size,
                              hipStream_t stream) {
  const float* q  = (const float*)d_in[0];
  const float* k  = (const float*)d_in[1];
  const float* v  = (const float*)d_in[2];
  const float* wq = (const float*)d_in[3];
  const float* bq = (const float*)d_in[4];
  const float* wk = (const float*)d_in[5];
  const float* bk = (const float*)d_in[6];
  const float* wv = (const float*)d_in[7];
  const float* bv = (const float*)d_in[8];
  const float* wo = (const float*)d_in[9];
  const float* bo = (const float*)d_in[10];

  const size_t HEAD_ELEMS = (size_t)BATCH * NHEAD * SEQ * DK;  // 4,194,304
  bf16_t* ws   = (bf16_t*)d_ws;
  bf16_t* qh   = ws;
  bf16_t* kh   = ws + HEAD_ELEMS;
  bf16_t* vh   = ws + 2 * HEAD_ELEMS;
  bf16_t* attn = ws + 3 * HEAD_ELEMS;

  dim3 gProj(BATCH * SEQ / 128, D_MODEL / 64, 3);  // (64, 8, 3)
  proj_qkv_kernel<<<gProj, 128, 0, stream>>>(q, k, v, wq, bq, wk, bk, wv, bv,
                                             qh, kh, vh);

  dim3 gAttn(SEQ / 64, BATCH * NHEAD);             // (64, 16)
  attn_kernel<<<gAttn, 128, 0, stream>>>(qh, kh, vh, attn);

  dim3 gOut(BATCH * SEQ / 128, D_MODEL / 64);      // (64, 8)
  proj_out_kernel<<<gOut, 128, 0, stream>>>(attn, wo, bo, (float*)d_out);
}
